// Attention_58153857187952
// MI455X (gfx1250) — matrix operand order repaired and checked by reading, not compiled, not run on hardware
//
#include <hip/hip_runtime.h>
#include <math.h>

typedef __bf16 bf16_t;
typedef __attribute__((ext_vector_type(16))) __bf16 v16bf;
typedef __attribute__((ext_vector_type(8)))  float   v8f;
typedef unsigned __attribute__((ext_vector_type(4))) u32x4;
typedef int      __attribute__((ext_vector_type(8))) i32x8;
typedef int      __attribute__((ext_vector_type(4))) i32x4;

#define BB     8
#define NSEQ   1024
#define DMODEL 1024
#define NH     16
#define HD     64
#define SCALE  0.125f        /* 64^-0.5 */
#define LN_EPS 1e-5f

union FragBF { v16bf v; uint4 q[2]; };

__device__ __forceinline__ v8f wmma_bf16(v16bf a, v16bf b, v8f c) {
  return __builtin_amdgcn_wmma_f32_16x16x32_bf16(false, a, false, b, (short)0, c,
                                                 false, false);
}

// order-preserving float <-> uint encoding for atomicMax on floats
__device__ __forceinline__ unsigned enc_f32(float f) {
  unsigned b = __float_as_uint(f);
  return (b & 0x80000000u) ? ~b : (b | 0x80000000u);
}
__device__ __forceinline__ float dec_f32(unsigned u) {
  unsigned b = (u & 0x80000000u) ? (u ^ 0x80000000u) : ~u;
  return __uint_as_float(b);
}

// ---------------------------------------------------------------------------
// TDM: DMA a 128-row x 32-col bf16 tile (row stride = rowStrideElems elems)
// from global into LDS at lds_byte_off, padding each 64B row with 16B so the
// LDS image has 40-element rows (bank-conflict-free frag reads).
// D# group0: count=1 | lds_addr | global_addr(57b) | type=2.
// D# group1: data_size=2B, pad_enable, pad_interval=16 DW, pad_amount=4 DW,
//            tensor_dim0=rowStride, tensor_dim1=big, tile=32x128, stride=K.
// Toolchain here uses the 6-arg builtin: (g0, g1, g2, g3, g4, cpol).
// ---------------------------------------------------------------------------
__device__ __forceinline__ void tdm_load_tile(unsigned lds_byte_off,
                                              const bf16_t* gsrc,
                                              int rowStrideElems) {
  unsigned long long ga = (unsigned long long)(size_t)gsrc;
  u32x4 g0;
  g0[0] = 1u;                                   // count=1 (valid user D#)
  g0[1] = lds_byte_off;                         // LDS destination (bytes)
  g0[2] = (unsigned)ga;                         // global_addr[31:0]
  g0[3] = (unsigned)((ga >> 32) & 0x01FFFFFFull) | 0x80000000u;  // [56:32]|type=2

  const unsigned tdim0 = (unsigned)rowStrideElems;   // elements per row
  const unsigned tdim1 = 1u << 20;                   // plenty of rows (no OOB)
  const unsigned tile0 = 32u, tile1 = 128u;
  const unsigned long long stride = (unsigned long long)rowStrideElems;
  i32x8 g1;
  g1[0] = (int)((1u << 16) | (1u << 20) | (3u << 22) | (3u << 25));
  g1[1] = (int)((tdim0 & 0xFFFFu) << 16);
  g1[2] = (int)(((tdim0 >> 16) & 0xFFFFu) | ((tdim1 & 0xFFFFu) << 16));
  g1[3] = (int)(((tdim1 >> 16) & 0xFFFFu) | (tile0 << 16));
  g1[4] = (int)tile1;
  g1[5] = (int)(unsigned)(stride & 0xFFFFFFFFull);
  g1[6] = (int)(unsigned)((stride >> 32) & 0xFFFFull);
  g1[7] = 0;
  i32x4 gz4 = {0, 0, 0, 0};
  i32x8 gz8 = {0, 0, 0, 0, 0, 0, 0, 0};
  __builtin_amdgcn_tensor_load_to_lds(g0, g1, gz4, gz4, gz8, 0);
}

__global__ void init_gmax_kernel(unsigned* g) {
  if (threadIdx.x == 0 && blockIdx.x == 0) *g = 0u;  // below enc of any float
}

__global__ void cvt_bf16_kernel(const float* __restrict__ in,
                                bf16_t* __restrict__ out, int n) {
  int i = blockIdx.x * blockDim.x + threadIdx.x;
  if (i < n) out[i] = (bf16_t)in[i];
}

// out[n*K + k] = in[k*N + n]  (store W transposed for contiguous B-frags)
__global__ void cvt_t_bf16_kernel(const float* __restrict__ in,
                                  bf16_t* __restrict__ out, int K, int N) {
  int i = blockIdx.x * blockDim.x + threadIdx.x;
  if (i < K * N) {
    int k = i / N, n = i - k * N;
    out[(size_t)n * K + k] = (bf16_t)in[i];
  }
}

// ---------------------------------------------------------------------------
// Tiled bf16 GEMM: C[M,N] = A[M,K] * Bt[N,K]^T, f32 accumulate in WMMA.
// Block tile 128x128, 8 waves (2m x 4n), each wave 64x32 (4x2 WMMA frags).
// Tiles staged into double-buffered LDS by the Tensor Data Mover (wave 0
// issues, TENSORcnt-tracked, overlapped with WMMA on the other buffer).
// MODE 0: store bf16 to head layout   [B,H,N,HD]      (Q, K)
// MODE 1: store bf16 to headT layout  [B,H,HD,N]      (V)
// MODE 2: outF = acc + bias[col] + res[row,col]  (f32)
// MODE 3: outB = gelu_exact(acc + bias[col])     (bf16)
// ---------------------------------------------------------------------------
template <int MODE>
__global__ __launch_bounds__(256)
void gemm_bf16_kernel(const bf16_t* __restrict__ A, const bf16_t* __restrict__ Bt,
                      const float* __restrict__ bias, const float* __restrict__ res,
                      float* __restrict__ outF, bf16_t* __restrict__ outB,
                      int M, int N, int K) {
  __shared__ __align__(16) bf16_t As[2][128][40];   // 128 x 32 + TDM pad
  __shared__ __align__(16) bf16_t Bs[2][128][40];
  const int tid  = threadIdx.x;
  const int lane = tid & 31, wave = tid >> 5;
  const int wm = wave & 1, wn = wave >> 1;        // 2 x 4 wave grid
  const int m0 = blockIdx.y * 128, n0 = blockIdx.x * 128;

  const v8f vzero = {0.f, 0.f, 0.f, 0.f, 0.f, 0.f, 0.f, 0.f};
  v8f acc[4][2];
#pragma unroll
  for (int i = 0; i < 4; i++)
#pragma unroll
    for (int j = 0; j < 2; j++) acc[i][j] = vzero;

  const int koff = (lane >> 4) * 8;    // A-frag K base within 16
  const int kb   = (lane >> 4) * 16;   // unused: B-frags use koff, same k order as A

  const bf16_t* Ablk = A  + (size_t)m0 * K;
  const bf16_t* Bblk = Bt + (size_t)n0 * K;
  if (wave == 0) {   // prologue DMA into buffer 0
    tdm_load_tile((unsigned)(size_t)&As[0][0][0], Ablk, K);
    tdm_load_tile((unsigned)(size_t)&Bs[0][0][0], Bblk, K);
  }

  int cur = 0;
  for (int k0 = 0; k0 < K; k0 += 32, cur ^= 1) {
    if (wave == 0) {
      if (k0 + 32 < K) {  // start next tile pair, then wait for current pair
        tdm_load_tile((unsigned)(size_t)&As[cur ^ 1][0][0], Ablk + k0 + 32, K);
        tdm_load_tile((unsigned)(size_t)&Bs[cur ^ 1][0][0], Bblk + k0 + 32, K);
        __builtin_amdgcn_s_wait_tensorcnt((short)2);
      } else {
        __builtin_amdgcn_s_wait_tensorcnt((short)0);
      }
    }
    __syncthreads();

    FragBF fa[4], fb[2];
#pragma unroll
    for (int mf = 0; mf < 4; mf++) {
      int r = wm * 64 + mf * 16 + (lane & 15);
      fa[mf].q[0] = *(const uint4*)&As[cur][r][koff];
      fa[mf].q[1] = *(const uint4*)&As[cur][r][16 + koff];
    }
#pragma unroll
    for (int nf = 0; nf < 2; nf++) {
      int r = wn * 32 + nf * 16 + (lane & 15);
      fb[nf].q[0] = *(const uint4*)&Bs[cur][r][koff];
      fb[nf].q[1] = *(const uint4*)&Bs[cur][r][16 + koff];
    }
#pragma unroll
    for (int mf = 0; mf < 4; mf++)
#pragma unroll
      for (int nf = 0; nf < 2; nf++)
        acc[mf][nf] = wmma_bf16(fa[mf].v, fb[nf].v, acc[mf][nf]);
    __syncthreads();
  }

  const int rsel = (lane >> 4) << 3;   // C-layout: lanes 16-31 hold M+8
  const int csel = lane & 15;
#pragma unroll
  for (int mf = 0; mf < 4; mf++) {
#pragma unroll
    for (int nf = 0; nf < 2; nf++) {
#pragma unroll
      for (int r = 0; r < 8; r++) {
        int row = m0 + wm * 64 + mf * 16 + rsel + r;
        int col = n0 + wn * 32 + nf * 16 + csel;
        float v = acc[mf][nf][r];
        if (MODE == 0) {
          int b = row >> 10, n = row & 1023, h = col >> 6, hd = col & 63;
          outB[((((size_t)b * NH + h) * NSEQ + n) << 6) + hd] = (bf16_t)v;
        } else if (MODE == 1) {
          int b = row >> 10, n = row & 1023, h = col >> 6, hd = col & 63;
          outB[(((size_t)b * NH + h) * HD + hd) * NSEQ + n] = (bf16_t)v;
        } else if (MODE == 2) {
          size_t idx = (size_t)row * N + col;
          outF[idx] = v + bias[col] + res[idx];
        } else {  // MODE 3: exact GELU
          float t = v + bias[col];
          float g = 0.5f * t * (1.0f + erff(t * 0.70710678118654752f));
          outB[(size_t)row * N + col] = (bf16_t)g;
        }
      }
    }
  }
}

// ---------------------------------------------------------------------------
// Pass 1: global max of scaled qk (pre-mask), WMMA tiles + atomicMax.
// Each wave owns a 16-row query strip of one (b,h); block = 128 rows.
// ---------------------------------------------------------------------------
__global__ __launch_bounds__(256)
void attn_max_kernel(const bf16_t* __restrict__ Q, const bf16_t* __restrict__ Kh,
                     unsigned* __restrict__ gmax) {
  const int tid = threadIdx.x, lane = tid & 31, wave = tid >> 5;
  const int bh = blockIdx.x >> 3;
  const int i0 = ((blockIdx.x & 7) << 7) + wave * 16;
  const int koff = (lane >> 4) * 8;
  const int kb   = (lane >> 4) * 16;
  const v8f vzero = {0.f, 0.f, 0.f, 0.f, 0.f, 0.f, 0.f, 0.f};

  const bf16_t* qrow = Q + ((size_t)bh * NSEQ + i0 + (lane & 15)) * HD;
  FragBF a0, a1;
  a0.q[0] = *(const uint4*)(qrow + koff);
  a0.q[1] = *(const uint4*)(qrow + 16 + koff);
  a1.q[0] = *(const uint4*)(qrow + 32 + koff);
  a1.q[1] = *(const uint4*)(qrow + 48 + koff);

  float wmax = -3.4e38f;
  for (int jb = 0; jb < NSEQ; jb += 16) {
    const bf16_t* krow = Kh + ((size_t)bh * NSEQ + jb + (lane & 15)) * HD;
    FragBF b0, b1;
    b0.q[0] = *(const uint4*)(krow + koff);
    b0.q[1] = *(const uint4*)(krow + 16 + koff);
    b1.q[0] = *(const uint4*)(krow + 32 + koff);
    b1.q[1] = *(const uint4*)(krow + 48 + koff);
    v8f s = vzero;
    s = wmma_bf16(a0.v, b0.v, s);
    s = wmma_bf16(a1.v, b1.v, s);
#pragma unroll
    for (int r = 0; r < 8; r++) wmax = fmaxf(wmax, s[r] * SCALE);
  }
  for (int off = 16; off > 0; off >>= 1)
    wmax = fmaxf(wmax, __shfl_xor(wmax, off, 32));
  if (lane == 0) atomicMax(gmax, enc_f32(wmax));
}

// ---------------------------------------------------------------------------
// Pass 2: flash-style attention faithful to the reference:
//   s = scale*qk + (keep_i&keep_j ? 0 : -gmax); denominator keeps masked
//   terms; numerator zeroed for masked. Online softmax, P staged via LDS
//   (C-layout -> A-layout), P*V with Vt read directly from global.
// ---------------------------------------------------------------------------
__global__ __launch_bounds__(256)
void attn_softmax_pv_kernel(const bf16_t* __restrict__ Q, const bf16_t* __restrict__ Kh,
                            const bf16_t* __restrict__ Vt,
                            const unsigned char* __restrict__ mask,
                            const unsigned* __restrict__ gmaxp,
                            bf16_t* __restrict__ ctx) {
  __shared__ float  S_lds[8][16][33];
  __shared__ __align__(16) bf16_t P_lds[8][16][40];
  __shared__ float rowM[8][16], rowL[8][16], rowA[8][16];

  const int tid = threadIdx.x, lane = tid & 31, w = tid >> 5;
  const int bh = blockIdx.x >> 3;
  const int b = bh >> 4, h = bh & 15;
  const int i0 = ((blockIdx.x & 7) << 7) + w * 16;
  const float negoff = -dec_f32(*gmaxp);
  const int koff = (lane >> 4) * 8;
  const int kb16 = (lane >> 4) * 16;
  const int rsel = (lane >> 4) << 3;
  const int csel = lane & 15;
  const v8f vzero = {0.f, 0.f, 0.f, 0.f, 0.f, 0.f, 0.f, 0.f};

  const bf16_t* qrow = Q + ((size_t)bh * NSEQ + i0 + csel) * HD;
  FragBF a0, a1;
  a0.q[0] = *(const uint4*)(qrow + koff);
  a0.q[1] = *(const uint4*)(qrow + 16 + koff);
  a1.q[0] = *(const uint4*)(qrow + 32 + koff);
  a1.q[1] = *(const uint4*)(qrow + 48 + koff);

  if (lane < 16) { rowM[w][lane] = -3.4e38f; rowL[w][lane] = 0.0f; }

  v8f o[4];
#pragma unroll
  for (int nf = 0; nf < 4; nf++) o[nf] = vzero;

  for (int jb = 0; jb < NSEQ; jb += 32) {
    // ---- scores: 16x32 tile of Q K^T
    v8f s[2];
    s[0] = vzero; s[1] = vzero;
#pragma unroll
    for (int nf = 0; nf < 2; nf++) {
      const bf16_t* krow = Kh + ((size_t)bh * NSEQ + jb + nf * 16 + csel) * HD;
      FragBF b0, b1;
      b0.q[0] = *(const uint4*)(krow + koff);
      b0.q[1] = *(const uint4*)(krow + 16 + koff);
      b1.q[0] = *(const uint4*)(krow + 32 + koff);
      b1.q[1] = *(const uint4*)(krow + 48 + koff);
      s[nf] = wmma_bf16(a0.v, b0.v, s[nf]);
      s[nf] = wmma_bf16(a1.v, b1.v, s[nf]);
    }
#pragma unroll
    for (int nf = 0; nf < 2; nf++)
#pragma unroll
      for (int r = 0; r < 8; r++)
        S_lds[w][rsel + r][nf * 16 + csel] = s[nf][r] * SCALE;
    __syncthreads();

    // ---- per-row online softmax update (lane < 16 owns one row)
    if (lane < 16) {
      int gi = i0 + lane;
      bool keep_i = (mask[b * NSEQ + gi] == 0);
      float m_old = rowM[w][lane];
      float mx = m_old;
      float sv[32];
#pragma unroll
      for (int c = 0; c < 32; c++) {
        bool keep_j = (mask[b * NSEQ + jb + c] == 0);
        float t = S_lds[w][lane][c] + ((keep_i && keep_j) ? 0.0f : negoff);
        sv[c] = t;
        mx = fmaxf(mx, t);
      }
      float alpha = __expf(m_old - mx);
      float l = rowL[w][lane] * alpha;
#pragma unroll
      for (int c = 0; c < 32; c++) {
        bool keep_j = (mask[b * NSEQ + jb + c] == 0);
        float p = __expf(sv[c] - mx);
        l += p;                                   // masked terms stay in denom
        P_lds[w][lane][c] = (bf16_t)((keep_i && keep_j) ? p : 0.0f);
      }
      rowM[w][lane] = mx; rowL[w][lane] = l; rowA[w][lane] = alpha;
    }
    __syncthreads();

    // ---- rescale running accumulators by per-row alpha
#pragma unroll
    for (int r = 0; r < 8; r++) {
      float al = rowA[w][rsel + r];
#pragma unroll
      for (int nf = 0; nf < 4; nf++) o[nf][r] *= al;
    }
    // ---- O += P (16x32) * V (32x64): A-frag from LDS, B-frags from Vt
    FragBF pa;
    pa.q[0] = *(const uint4*)&P_lds[w][csel][koff];
    pa.q[1] = *(const uint4*)&P_lds[w][csel][16 + koff];
#pragma unroll
    for (int nf = 0; nf < 4; nf++) {
      const bf16_t* vrow =
          Vt + ((size_t)bh * HD + nf * 16 + csel) * NSEQ + jb + koff;
      FragBF vb;
      vb.q[0] = *(const uint4*)(vrow);
      vb.q[1] = *(const uint4*)(vrow + 16);
      o[nf] = wmma_bf16(pa.v, vb.v, o[nf]);
    }
    __syncthreads();
  }

  // ---- normalize and store context in [B,N,D] bf16 (row-major for next GEMM)
#pragma unroll
  for (int r = 0; r < 8; r++) {
    float linv = 1.0f / rowL[w][rsel + r];
    int gi = i0 + rsel + r;
#pragma unroll
    for (int nf = 0; nf < 4; nf++) {
      int d = nf * 16 + csel;
      ctx[((size_t)b * NSEQ + gi) * DMODEL + h * HD + d] = (bf16_t)(o[nf][r] * linv);
    }
  }
}

// ---------------------------------------------------------------------------
// LayerNorm over D=1024; one block per row; dual f32 + bf16 outputs.
// ---------------------------------------------------------------------------
__global__ __launch_bounds__(256)
void layernorm_kernel(const float* __restrict__ x, const float* __restrict__ g,
                      const float* __restrict__ b, float* __restrict__ oF,
                      bf16_t* __restrict__ oB) {
  __shared__ float s1[256], s2[256];
  const int row = blockIdx.x, tid = threadIdx.x;
  const float* p = x + (size_t)row * DMODEL;
  float v[4], sum = 0.f, sq = 0.f;
#pragma unroll
  for (int i = 0; i < 4; i++) {
    v[i] = p[tid + i * 256];
    sum += v[i];
    sq += v[i] * v[i];
  }
  s1[tid] = sum; s2[tid] = sq;
  __syncthreads();
  for (int s = 128; s > 0; s >>= 1) {
    if (tid < s) { s1[tid] += s1[tid + s]; s2[tid] += s2[tid + s]; }
    __syncthreads();
  }
  float mean = s1[0] * (1.0f / DMODEL);
  float var  = s2[0] * (1.0f / DMODEL) - mean * mean;
  float rstd = rsqrtf(var + LN_EPS);
#pragma unroll
  for (int i = 0; i < 4; i++) {
    int c = tid + i * 256;
    float y = (v[i] - mean) * rstd * g[c] + b[c];
    oF[(size_t)row * DMODEL + c] = y;
    if (oB) oB[(size_t)row * DMODEL + c] = (bf16_t)y;
  }
}

// ---------------------------------------------------------------------------
extern "C" void kernel_launch(void* const* d_in, const int* in_sizes, int n_in,
                              void* d_out, int out_size, void* d_ws, size_t ws_size,
                              hipStream_t stream) {
  (void)in_sizes; (void)n_in; (void)out_size; (void)ws_size;
  const float* src      = (const float*)d_in[0];
  const float* wq       = (const float*)d_in[1];
  const float* wk       = (const float*)d_in[2];
  const float* wv       = (const float*)d_in[3];
  const float* wc       = (const float*)d_in[4];
  const float* b_concat = (const float*)d_in[5];
  const float* w_ffn1   = (const float*)d_in[6];
  const float* b_ffn1   = (const float*)d_in[7];
  const float* w_ffn2   = (const float*)d_in[8];
  const float* b_ffn2   = (const float*)d_in[9];
  const float* ln1_g    = (const float*)d_in[10];
  const float* ln1_b    = (const float*)d_in[11];
  const float* ln2_g    = (const float*)d_in[12];
  const float* ln2_b    = (const float*)d_in[13];
  const unsigned char* mask = (const unsigned char*)d_in[14];
  float* out = (float*)d_out;

  char* ws = (char*)d_ws;
  size_t off = 0;
  auto walloc = [&](size_t bytes) -> void* {
    void* p = ws + off;
    off = (off + bytes + 255) & ~(size_t)255;
    return p;
  };
  const size_t TOK = (size_t)BB * NSEQ;  // 8192 tokens

  bf16_t* srcB = (bf16_t*)walloc(TOK * DMODEL * 2);
  bf16_t* wqT  = (bf16_t*)walloc((size_t)DMODEL * DMODEL * 2);
  bf16_t* wkT  = (bf16_t*)walloc((size_t)DMODEL * DMODEL * 2);
  bf16_t* wvT  = (bf16_t*)walloc((size_t)DMODEL * DMODEL * 2);
  bf16_t* wcT  = (bf16_t*)walloc((size_t)DMODEL * DMODEL * 2);
  bf16_t* w1T  = (bf16_t*)walloc((size_t)DMODEL * 4 * DMODEL * 2);
  bf16_t* w2T  = (bf16_t*)walloc((size_t)DMODEL * 4 * DMODEL * 2);
  bf16_t* Qh   = (bf16_t*)walloc(TOK * DMODEL * 2);
  bf16_t* Kh   = (bf16_t*)walloc(TOK * DMODEL * 2);
  bf16_t* VtB  = (bf16_t*)walloc(TOK * DMODEL * 2);
  bf16_t* ctx  = (bf16_t*)walloc(TOK * DMODEL * 2);
  float*  y1   = (float*)walloc(TOK * DMODEL * 4);
  float*  xF   = (float*)walloc(TOK * DMODEL * 4);
  bf16_t* xB   = (bf16_t*)walloc(TOK * DMODEL * 2);
  bf16_t* hB   = (bf16_t*)walloc(TOK * 4 * DMODEL * 2);
  float*  y2   = (float*)walloc(TOK * DMODEL * 4);
  unsigned* gmax = (unsigned*)walloc(256);

  dim3 blk(256);
  init_gmax_kernel<<<1, 1, 0, stream>>>(gmax);

  {
    int n = (int)(TOK * DMODEL);
    cvt_bf16_kernel<<<(n + 255) / 256, blk, 0, stream>>>(src, srcB, n);
  }
  {
    int n = DMODEL * DMODEL;
    int g = (n + 255) / 256;
    cvt_t_bf16_kernel<<<g, blk, 0, stream>>>(wq, wqT, DMODEL, DMODEL);
    cvt_t_bf16_kernel<<<g, blk, 0, stream>>>(wk, wkT, DMODEL, DMODEL);
    cvt_t_bf16_kernel<<<g, blk, 0, stream>>>(wv, wvT, DMODEL, DMODEL);
    cvt_t_bf16_kernel<<<g, blk, 0, stream>>>(wc, wcT, DMODEL, DMODEL);
  }
  {
    int n = DMODEL * 4 * DMODEL;
    int g = (n + 255) / 256;
    cvt_t_bf16_kernel<<<g, blk, 0, stream>>>(w_ffn1, w1T, DMODEL, 4 * DMODEL);
    cvt_t_bf16_kernel<<<g, blk, 0, stream>>>(w_ffn2, w2T, 4 * DMODEL, DMODEL);
  }

  dim3 g_sq(DMODEL / 128, (unsigned)(TOK / 128));       // 8 x 64
  gemm_bf16_kernel<0><<<g_sq, blk, 0, stream>>>(srcB, wqT, nullptr, nullptr,
                                                nullptr, Qh, (int)TOK, DMODEL, DMODEL);
  gemm_bf16_kernel<0><<<g_sq, blk, 0, stream>>>(srcB, wkT, nullptr, nullptr,
                                                nullptr, Kh, (int)TOK, DMODEL, DMODEL);
  gemm_bf16_kernel<1><<<g_sq, blk, 0, stream>>>(srcB, wvT, nullptr, nullptr,
                                                nullptr, VtB, (int)TOK, DMODEL, DMODEL);

  const int attn_blocks = BB * NH * (NSEQ / 128);       // 1024
  attn_max_kernel<<<attn_blocks, blk, 0, stream>>>(Qh, Kh, gmax);
  attn_softmax_pv_kernel<<<attn_blocks, blk, 0, stream>>>(Qh, Kh, VtB, mask, gmax, ctx);

  gemm_bf16_kernel<2><<<g_sq, blk, 0, stream>>>(ctx, wcT, b_concat, src,
                                                y1, nullptr, (int)TOK, DMODEL, DMODEL);
  layernorm_kernel<<<(unsigned)TOK, blk, 0, stream>>>(y1, ln1_g, ln1_b, xF, xB);

  dim3 g_f1(4 * DMODEL / 128, (unsigned)(TOK / 128));   // 32 x 64
  gemm_bf16_kernel<3><<<g_f1, blk, 0, stream>>>(xB, w1T, b_ffn1, nullptr,
                                                nullptr, hB, (int)TOK, 4 * DMODEL, DMODEL);
  gemm_bf16_kernel<2><<<g_sq, blk, 0, stream>>>(hB, w2T, b_ffn2, xF,
                                                y2, nullptr, (int)TOK, DMODEL, 4 * DMODEL);
  layernorm_kernel<<<(unsigned)TOK, blk, 0, stream>>>(y2, ln2_g, ln2_b, out, nullptr);
}
